// CTR_Model_HAW_80083960201470
// MI455X (gfx1250) — matrix operand order repaired and checked by reading, not compiled, not run on hardware
//
#include <hip/hip_runtime.h>
#include <math.h>

#define NT    64000
#define EDG   400000
#define BB    128
#define LL    50
#define KK    4
#define DD    64
#define NUSER 100000

typedef __attribute__((ext_vector_type(16))) _Float16 v16h;
typedef __attribute__((ext_vector_type(8)))  float    v8f;
typedef __attribute__((ext_vector_type(4)))  int      v4i;
typedef __attribute__((address_space(1))) v4i v4i_g;   // global
typedef __attribute__((address_space(3))) v4i v4i_l;   // LDS

// ---------------- wave32 reduction ----------------
__device__ __forceinline__ float wred32(float v) {
#pragma unroll
  for (int off = 16; off > 0; off >>= 1) v += __shfl_xor(v, off, 32);
  return v;
}

// ---------------- threefry2x32 (JAX key(1)) ----------------
__device__ __forceinline__ void tf_round(unsigned& x0, unsigned& x1, int r) {
  x0 += x1;
  x1 = (x1 << r) | (x1 >> (32 - r));
  x1 ^= x0;
}
__device__ __forceinline__ unsigned threefry_bits(int i) {
  const int half = (BB * LL * KK) / 2;  // 12800
  unsigned c0 = (i < half) ? (unsigned)i : (unsigned)(i - half);
  unsigned c1 = c0 + (unsigned)half;
  const unsigned ks0 = 0u, ks1 = 1u, ks2 = 0x1BD11BDAu ^ 0u ^ 1u;
  unsigned x0 = c0 + ks0, x1 = c1 + ks1;
  const int ra[4] = {13, 15, 26, 6}, rb[4] = {17, 29, 16, 24};
#pragma unroll
  for (int j = 0; j < 4; ++j) tf_round(x0, x1, ra[j]);
  x0 += ks1; x1 += ks2 + 1u;
#pragma unroll
  for (int j = 0; j < 4; ++j) tf_round(x0, x1, rb[j]);
  x0 += ks2; x1 += ks0 + 2u;
#pragma unroll
  for (int j = 0; j < 4; ++j) tf_round(x0, x1, ra[j]);
  x0 += ks0; x1 += ks1 + 3u;
#pragma unroll
  for (int j = 0; j < 4; ++j) tf_round(x0, x1, rb[j]);
  x0 += ks1; x1 += ks2 + 4u;
#pragma unroll
  for (int j = 0; j < 4; ++j) tf_round(x0, x1, ra[j]);
  x0 += ks2; x1 += ks0 + 5u;
  return (i < half) ? x0 : x1;
}
__device__ __forceinline__ float gumbel_at(int i) {
  unsigned bits = threefry_bits(i);
  float f = __uint_as_float((bits >> 9) | 0x3F800000u) - 1.0f;  // [0,1)
  const float eps = 1e-4f;
  float u = f * (1.0f - 2.0f * eps) + eps;
  u = fmaxf(eps, u);
  return -logf(-logf(u));
}

// ---------------- utility kernels ----------------
__global__ void k_fill_f(float* __restrict__ p, float v, int n) {
  int i = blockIdx.x * blockDim.x + threadIdx.x;
  if (i < n) p[i] = v;
}
__global__ void k_zero_f4(float4* __restrict__ p, int n4) {
  int i = blockIdx.x * blockDim.x + threadIdx.x;
  if (i < n4) p[i] = make_float4(0.f, 0.f, 0.f, 0.f);
}
__global__ void k_degree(const int* __restrict__ ei, float* __restrict__ deg) {
  int e = blockIdx.x * blockDim.x + threadIdx.x;
  if (e < EDG) atomicAdd(&deg[ei[EDG + e]], 1.0f);
}
__global__ void k_dinv(const float* __restrict__ deg, float* __restrict__ dinv) {
  int i = blockIdx.x * blockDim.x + threadIdx.x;
  if (i < NT) { float d = deg[i]; dinv[i] = (d > 0.f) ? rsqrtf(d) : 0.f; }
}
__global__ void k_base(const int* __restrict__ ei, const float* __restrict__ dinv,
                       float* __restrict__ base) {
  int e = blockIdx.x * blockDim.x + threadIdx.x;
  if (e < EDG) base[e] = dinv[ei[e]] * dinv[ei[EDG + e]];
}
__global__ void k_gather(const float4* __restrict__ embed4, const int* __restrict__ node_ids,
                         float4* __restrict__ x4) {
  int i = blockIdx.x * blockDim.x + threadIdx.x;  // NT*64 float4's
  if (i < NT * 64) {
    int n = i >> 6, j = i & 63;
    x4[i] = embed4[(long)node_ids[n] * 64 + j];
  }
}

// ---------------- per-edge routing weights: w[e][k] = base[e]*softmax_k(s[e]) ----
__global__ void k_w(const float* __restrict__ s, const float* __restrict__ base,
                    float* __restrict__ wq) {
  int e = blockIdx.x * blockDim.x + threadIdx.x;
  if (e >= EDG) return;
  float s0 = s[e * 4 + 0], s1 = s[e * 4 + 1], s2 = s[e * 4 + 2], s3 = s[e * 4 + 3];
  float mx = fmaxf(fmaxf(s0, s1), fmaxf(s2, s3));
  float e0 = expf(s0 - mx), e1 = expf(s1 - mx), e2 = expf(s2 - mx), e3 = expf(s3 - mx);
  float inv = base[e] / (e0 + e1 + e2 + e3);
  wq[e * 4 + 0] = e0 * inv;
  wq[e * 4 + 1] = e1 * inv;
  wq[e * 4 + 2] = e2 * inv;
  wq[e * 4 + 3] = e3 * inv;
}

// ---------------- graph conv: message scatter (L2-resident atomics) ----------------
__global__ void k_msg(const float* __restrict__ xs, float* __restrict__ xd,
                      const float* __restrict__ wq, const int* __restrict__ ei) {
  const int gid = blockIdx.x * blockDim.x + threadIdx.x;  // EDG*64 threads
  const int e = gid >> 6;
  if (e >= EDG) return;
  const int d = gid & 63;
  const int row = ei[e], col = ei[EDG + e];
  const float w0 = wq[e * 4 + 0], w1 = wq[e * 4 + 1], w2 = wq[e * 4 + 2], w3 = wq[e * 4 + 3];
  const float* xr = xs + (long)row * 256 + d;
  float* xw = xd + (long)col * 256 + d;
  atomicAdd(xw + 0,   xr[0]   * w0);
  atomicAdd(xw + 64,  xr[64]  * w1);
  atomicAdd(xw + 128, xr[128] * w2);
  atomicAdd(xw + 192, xr[192] * w3);
}

// ---------------- graph conv: tanh-gated edge score, s += z ----------------
__global__ void k_edge(const float* __restrict__ x, float* __restrict__ s,
                       const int* __restrict__ ei) {
  const int wid = (blockIdx.x * blockDim.x + threadIdx.x) >> 5;  // one wave32 per edge
  if (wid >= EDG) return;
  const int lane = threadIdx.x & 31;
  const int row = ei[wid], col = ei[EDG + wid];
  const float* xr = x + (long)row * 256;
  const float* xc = x + (long)col * 256;
#pragma unroll
  for (int k = 0; k < 4; ++k) {
    float p = xr[k * 64 + lane]      * tanhf(xc[k * 64 + lane]) +
              xr[k * 64 + lane + 32] * tanhf(xc[k * 64 + lane + 32]);
    p = wred32(p);
    if (lane == 0) s[wid * 4 + k] += p;
  }
}

// ---------------- WMMA GEMM: out[m,:] = A[m,:]*W^T + residual ----------------
// mode 0: m=(b,l,k), A row = x[seq_map[b,l]][k], resi = embed[node_ids[seq_map]], mask l<len
// mode 1: m=(b,k),   A row = x[tar_map[b]][k],   resi = embed[sid[b]+NUSER]
__global__ void k_gemm(const float* __restrict__ x, const float* __restrict__ W,
                       const float* __restrict__ embed, const int* __restrict__ map,
                       const int* __restrict__ node_ids, const int* __restrict__ sid,
                       const int* __restrict__ seq_len, float* __restrict__ outb,
                       int mode) {
  __shared__ float    Wf[64 * 64];
  __shared__ _Float16 Wh[64 * 64];
  const int tid = threadIdx.x;

#if __has_builtin(__builtin_amdgcn_global_load_async_to_lds_b128)
  // Stage W (16 KB) global->LDS via gfx1250 async DMA (ASYNCcnt), no VGPR round-trip.
  {
#pragma unroll
    for (int it = 0; it < 4; ++it) {
      __builtin_amdgcn_global_load_async_to_lds_b128(
          (v4i_g*)(W + it * 1024 + tid * 4),
          (v4i_l*)(Wf + it * 1024 + tid * 4),
          0, 0);
    }
    asm volatile("s_wait_asynccnt 0" ::: "memory");
  }
  __syncthreads();
#pragma unroll 4
  for (int i = tid; i < 4096; i += 256) Wh[i] = (_Float16)Wf[i];
#else
#pragma unroll 4
  for (int i = tid; i < 4096; i += 256) { Wf[i] = W[i]; Wh[i] = (_Float16)W[i]; }
#endif
  __syncthreads();

  const int lane = tid & 31;
  const int wv = tid >> 5;
  const int mtile = blockIdx.x * 8 + wv;

  // ---- A fragment (16-bit A 16x32, contiguous k: lanes 0-15 K=0..15, 16-31 K=16..31) ----
  const int r  = lane & 15;
  const int lo = (lane >> 4) ? 16 : 0;
  const int m = mtile * 16 + r;
  int kch, node;
  if (mode == 0) {
    int b = m / (LL * KK); int rem = m % (LL * KK);
    int l = rem >> 2; kch = rem & 3;
    node = map[b * LL + l];
  } else {
    int b = m >> 2; kch = m & 3;
    node = map[b];
  }
  const float* Arow = x + (long)node * 256 + kch * 64;

  v16h afrag[2];
#pragma unroll
  for (int kt = 0; kt < 2; ++kt) {
#pragma unroll
    for (int j = 0; j < 8; ++j) {
      afrag[kt][j]     = (_Float16)Arow[kt * 32 + lo + j];
      afrag[kt][8 + j] = (_Float16)Arow[kt * 32 + lo + 8 + j];
    }
  }

  // ---- 4 N-tiles x 2 K-tiles of v_wmma_f32_16x16x32_f16 ----
  const int kb = (lane >> 4) ? 16 : 0;  // B 32x16 layout: lanes 0-15 K=0..15, 16-31 K=16..31
  v8f acc[4] = {};
#pragma unroll
  for (int nt = 0; nt < 4; ++nt) {
    const int e = nt * 16 + r;
#pragma unroll
    for (int kt = 0; kt < 2; ++kt) {
      v16h bfrag;
#pragma unroll
      for (int j = 0; j < 16; ++j) bfrag[j] = Wh[e * 64 + kt * 32 + kb + j];
      acc[nt] = __builtin_amdgcn_wmma_f32_16x16x32_f16(
          false, afrag[kt], false, bfrag, (short)0, acc[nt], false, false);
    }
  }

  // ---- store (f32 C/D 16x16 layout: VGPR rr -> M=rr (lanes<16) / rr+8) ----
  const int nc = lane & 15;
  const int mrow_off = (lane >> 4) ? 8 : 0;
#pragma unroll
  for (int rr = 0; rr < 8; ++rr) {
    const int mg = mtile * 16 + rr + mrow_off;
    long resiBase; bool valid = true;
    if (mode == 0) {
      int b = mg / (LL * KK); int rem = mg % (LL * KK);
      int l = rem >> 2; int k2 = rem & 3;
      int nd = map[b * LL + l];
      resiBase = (long)node_ids[nd] * 256 + k2 * 64;
      valid = (l < seq_len[b]);
    } else {
      int b = mg >> 2; int k2 = mg & 3;
      resiBase = (long)(sid[b] + NUSER) * 256 + k2 * 64;
    }
#pragma unroll
    for (int nt = 0; nt < 4; ++nt) {
      const int e = nt * 16 + nc;
      float v = acc[nt][rr] + embed[resiBase + e];
      outb[(long)mg * 64 + e] = valid ? v : 0.0f;
    }
  }
}

// ---------------- final scoring: one wave32 per batch row ----------------
__global__ void k_score(const float* __restrict__ embed, const float* __restrict__ aV,
                        const float* __restrict__ seqE, const float* __restrict__ delta,
                        const float* __restrict__ tau_p, const float* __restrict__ seqs_time,
                        const float* __restrict__ clk_time, const int* __restrict__ uid,
                        const int* __restrict__ seq_idx, const int* __restrict__ seq_len,
                        float* __restrict__ out) {
  const int b = blockIdx.x * 8 + (threadIdx.x >> 5);
  if (b >= BB) return;
  const int lane = threadIdx.x & 31;
  const int u = uid[b];
  const float* au = embed + (long)u * 256;
  const float* av = aV + (long)b * 256;

  float au0[4], au1[4], av0[4], av1[4], nu[4], nv[4], acc[4];
#pragma unroll
  for (int k = 0; k < 4; ++k) {
    au0[k] = au[k * 64 + lane]; au1[k] = au[k * 64 + lane + 32];
    av0[k] = av[k * 64 + lane]; av1[k] = av[k * 64 + lane + 32];
    float duu = wred32(au0[k] * au0[k] + au1[k] * au1[k]);
    float dvv = wred32(av0[k] * av0[k] + av1[k] * av1[k]);
    float duv = wred32(au0[k] * av0[k] + au1[k] * av1[k]);
    nu[k] = fmaxf(sqrtf(duu), 1e-8f);
    nv[k] = fmaxf(sqrtf(dvv), 1e-8f);
    acc[k] = duv / (nu[k] * nv[k]);  // mu_u_v
  }

  const int len = seq_len[b];
  const float du = delta[u];
  const float ct = clk_time[b];
  for (int l = 0; l < len; ++l) {
    const float* se = seqE + (long)((b * LL + l) * 4) * 64;
    float gamma[4], tk[4];
#pragma unroll
    for (int k = 0; k < 4; ++k) {
      float s0 = se[k * 64 + lane], s1 = se[k * 64 + lane + 32];
      float dss = wred32(s0 * s0 + s1 * s1);
      float dsv = wred32(s0 * av0[k] + s1 * av1[k]);
      float dsu = wred32(s0 * au0[k] + s1 * au1[k]);
      float ns = fmaxf(sqrtf(dss), 1e-8f);
      gamma[k] = dsv / (ns * nv[k]);
      float logit = dsu / (ns * nu[k]);
      tk[k] = logit + gumbel_at((b * LL + l) * 4 + k);
    }
    float itau = 1.0f / tau_p[seq_idx[b * LL + l]];
    float mx = fmaxf(fmaxf(tk[0], tk[1]), fmaxf(tk[2], tk[3]));
    float p[4], psum = 0.f;
#pragma unroll
    for (int k = 0; k < 4; ++k) { p[k] = expf((tk[k] - mx) * itau); psum += p[k]; }
    float j = expf(-(ct - seqs_time[b * LL + l]) * (1.0f / 1000.0f) * du);
    float jp = j / psum;
#pragma unroll
    for (int k = 0; k < 4; ++k) acc[k] += jp * gamma[k] * p[k];
  }
  if (lane == 0) out[b] = 0.25f * (acc[0] + acc[1] + acc[2] + acc[3]);
}

// ---------------- host ----------------
extern "C" void kernel_launch(void* const* d_in, const int* in_sizes, int n_in,
                              void* d_out, int out_size, void* d_ws, size_t ws_size,
                              hipStream_t stream) {
  const float* embed     = (const float*)d_in[0];
  const float* delta     = (const float*)d_in[1];
  const float* tau_p     = (const float*)d_in[2];
  const float* W1        = (const float*)d_in[3];
  const float* W2        = (const float*)d_in[4];
  const float* seqs_time = (const float*)d_in[5];
  const float* clk_time  = (const float*)d_in[6];
  const int*   uid       = (const int*)d_in[7];
  const int*   sid       = (const int*)d_in[8];
  const int*   node_ids  = (const int*)d_in[9];
  const int*   ei        = (const int*)d_in[10];
  const int*   seq_map   = (const int*)d_in[11];
  const int*   tar_map   = (const int*)d_in[12];
  const int*   seq_idx   = (const int*)d_in[13];
  const int*   seq_len   = (const int*)d_in[14];
  float* out = (float*)d_out;
  (void)in_sizes; (void)n_in; (void)out_size; (void)ws_size;

  char* w = (char*)d_ws;
  auto alloc = [&](size_t bytes) {
    char* p = w;
    w += (bytes + 255) & ~(size_t)255;
    return p;
  };
  float* xA   = (float*)alloc((size_t)NT * 256 * 4);
  float* xB   = (float*)alloc((size_t)NT * 256 * 4);
  float* deg  = (float*)alloc((size_t)NT * 4);
  float* dinv = (float*)alloc((size_t)NT * 4);
  float* base = (float*)alloc((size_t)EDG * 4);
  float* s    = (float*)alloc((size_t)EDG * 4 * 4);
  float* wq   = (float*)alloc((size_t)EDG * 4 * 4);
  float* seqE = (float*)alloc((size_t)BB * LL * KK * 64 * 4);
  float* aV   = (float*)alloc((size_t)BB * KK * 64 * 4);

  // degree normalization + init
  k_fill_f<<<(NT + 255) / 256, 256, 0, stream>>>(deg, 0.0f, NT);
  k_fill_f<<<(EDG * 4 + 255) / 256, 256, 0, stream>>>(s, 1.0f, EDG * 4);
  k_degree<<<(EDG + 255) / 256, 256, 0, stream>>>(ei, deg);
  k_dinv<<<(NT + 255) / 256, 256, 0, stream>>>(deg, dinv);
  k_base<<<(EDG + 255) / 256, 256, 0, stream>>>(ei, dinv, base);

  // x = embed[node_ids]
  k_gather<<<(NT * 64 + 255) / 256, 256, 0, stream>>>((const float4*)embed, node_ids,
                                                      (float4*)xA);

  // two conv layers, ping-pong xA/xB (final x lands back in xA)
  float* xs = xA;
  float* xd = xB;
  for (int t = 0; t < 2; ++t) {
    k_w<<<(EDG + 255) / 256, 256, 0, stream>>>(s, base, wq);
    k_zero_f4<<<(NT * 64 + 255) / 256, 256, 0, stream>>>((float4*)xd, NT * 64);
    k_msg<<<(EDG * 64 + 255) / 256, 256, 0, stream>>>(xs, xd, wq, ei);
    k_edge<<<(EDG * 32 + 255) / 256, 256, 0, stream>>>(xd, s, ei);
    float* tmp = xs; xs = xd; xd = tmp;
  }

  // WMMA GEMMs: seq_embed (25600x64 rows) and a_v (512 rows)
  k_gemm<<<(BB * LL * KK / 16) / 8, 256, 0, stream>>>(xs, W1, embed, seq_map, node_ids,
                                                      nullptr, seq_len, seqE, 0);
  k_gemm<<<(BB * KK / 16) / 8, 256, 0, stream>>>(xs, W2, embed, tar_map, nullptr,
                                                 sid, nullptr, aV, 1);

  // final scoring
  k_score<<<(BB + 7) / 8, 256, 0, stream>>>(embed, aV, seqE, delta, tau_p, seqs_time,
                                            clk_time, uid, seq_idx, seq_len, out);
}
